// FuseNet_24644522344854
// MI455X (gfx1250) — compile-verified, matrix operand order checked by reading, not run on hardware
//
#include <hip/hip_runtime.h>

#define Bb 2
#define Hh 256
#define Ww 512
#define Nn 8192
#define Kk 9
#define CS 64          // channel stride for all f16 activation buffers
#define Hh2 128
#define Ww2 256
#define EPSF 1e-5f

typedef _Float16 v16h __attribute__((ext_vector_type(16)));
typedef _Float16 v8h  __attribute__((ext_vector_type(8)));
typedef float    v8f  __attribute__((ext_vector_type(8)));
typedef unsigned int v4u __attribute__((ext_vector_type(4)));
typedef int          v4i __attribute__((ext_vector_type(4)));
typedef int          v8i __attribute__((ext_vector_type(8)));

union U16 { v16h v; v8h h[2]; _Float16 e[16]; };

// ---- Tensor Data Mover availability (clang-22: 5-arg builtin, clang-23: 6-arg) ----
#if defined(__HIP_DEVICE_COMPILE__) && defined(__has_builtin)
#if __has_builtin(__builtin_amdgcn_tensor_load_to_lds)
#define USE_TDM 1
#endif
#endif

#ifdef USE_TDM
// 1-D DMA of `ndwords` dwords from global `src` to LDS offset `lds_off`.
// D# per CDNA5 ISA: group0 = {count=1 | lds_addr | global_addr | type=2},
// group1 = {data_size=4B, tensor_dim0=ndwords, tensor_dim1=1, tile_dim0=ndwords,
//           tensor_dim0_stride=ndwords}.  ndwords must be <= 65535.
__device__ __forceinline__ void tdm_load_1d(const void* src, unsigned ndwords, unsigned lds_off){
  unsigned long long ga = (unsigned long long)src;
  v4u g0;
  g0[0] = 1u;                                            // count=1, user descriptor
  g0[1] = lds_off;                                       // lds_addr (bytes)
  g0[2] = (unsigned)(ga & 0xffffffffu);                  // global_addr[31:0]
  g0[3] = (unsigned)((ga >> 32) & 0x01ffffffu) | (2u << 30);  // addr[56:32] | type=2
  v8i g1;
  g1[0] = (int)(2u << 16);                               // data_size: 2 -> 4 bytes
  g1[1] = (int)((ndwords & 0xffffu) << 16);              // tensor_dim0[15:0]
  g1[2] = (int)((1u << 16) | (ndwords >> 16));           // tensor_dim1=1 | tensor_dim0[31:16]
  g1[3] = (int)((ndwords & 0xffffu) << 16);              // tile_dim0 | tensor_dim1[31:16]=0
  g1[4] = 0;                                             // tile_dim1/2 unused
  g1[5] = (int)ndwords;                                  // tensor_dim0_stride[31:0]
  g1[6] = 0;
  g1[7] = 0;
  v4i z4 = {0,0,0,0};
#if __clang_major__ >= 23
  v8i z8 = {0,0,0,0,0,0,0,0};
  __builtin_amdgcn_tensor_load_to_lds(g0, g1, z4, z4, z8, 0);
#else
  __builtin_amdgcn_tensor_load_to_lds(g0, g1, z4, z4, 0);
#endif
}
#endif

// stage `nh` halves of weights from global wp into ldsw (LDS offset 0); whole block participates
__device__ __forceinline__ void stage_weights(_Float16* ldsw, const _Float16* wp, int nh){
  __syncthreads();
#ifdef USE_TDM
  if ((threadIdx.x >> 5) == 0){                          // one wave issues the DMA
    tdm_load_1d(wp, (unsigned)(nh >> 1), 0u);            // ldsw is the only LDS object -> offset 0
    __builtin_amdgcn_s_wait_tensorcnt(0);
  }
#else
  for (int i = threadIdx.x; i < nh; i += 256) ldsw[i] = wp[i];
#endif
  __syncthreads();
}

// ---------------------------------------------------------------- pix_idx
__global__ void k_pix_idx(const unsigned char* __restrict__ mask, int* __restrict__ pix){
  int b = blockIdx.x;
  const unsigned char* m = mask + (size_t)b * Hh * Ww;
  int cnt = 0;
  for (int i = 0; i < Hh * Ww; ++i){
    if (m[i] && cnt < Nn){ pix[b * Nn + cnt] = i; ++cnt; }
  }
  while (cnt < Nn){ pix[b * Nn + cnt] = 0; ++cnt; }
}

// ---------------------------------------------------------------- zero fill (f16)
__global__ void k_zero_h(_Float16* __restrict__ p, size_t n){
  size_t i = (size_t)blockIdx.x * blockDim.x + threadIdx.x;
  size_t st = (size_t)gridDim.x * blockDim.x;
  for (; i < n; i += st) p[i] = (_Float16)0.f;
}

// ---------------------------------------------------------------- stems
__global__ void k_sparse1(const float* __restrict__ sd, const float* __restrict__ w,
                          const float* __restrict__ bc, const float* __restrict__ g,
                          const float* __restrict__ bb, _Float16* __restrict__ out){
  int idx = blockIdx.x * blockDim.x + threadIdx.x;
  if (idx >= Bb * Hh2 * Ww2 * 16) return;
  int c = idx & 15; int q = idx >> 4;
  int wo = q % Ww2; q /= Ww2;
  int ho = q % Hh2; int b = q / Hh2;
  float acc = 0.f;
  for (int i = 0; i < 3; ++i){
    int hi = 2*ho - 1 + i; if ((unsigned)hi >= Hh) continue;
    for (int j = 0; j < 3; ++j){
      int wi = 2*wo - 1 + j; if ((unsigned)wi >= Ww) continue;
      acc += sd[((size_t)b*Hh + hi)*Ww + wi] * w[c*9 + i*3 + j];
    }
  }
  float v = (acc + bc[c]) * (g[c]*rsqrtf(1.f+EPSF)) + bb[c];
  out[idx] = (_Float16)fmaxf(v, 0.f);
}

__global__ void k_rgbd1(const float* __restrict__ img, const float* __restrict__ sd,
                        const float* __restrict__ w, const float* __restrict__ bc,
                        const float* __restrict__ g, const float* __restrict__ bb,
                        _Float16* __restrict__ out){
  int idx = blockIdx.x * blockDim.x + threadIdx.x;
  if (idx >= Bb * Hh2 * Ww2 * 32) return;
  int c = idx & 31; int q = idx >> 5;
  int wo = q % Ww2; q /= Ww2;
  int ho = q % Hh2; int b = q / Hh2;
  float acc = 0.f;
  for (int i = 0; i < 3; ++i){
    int hi = 2*ho - 1 + i; if ((unsigned)hi >= Hh) continue;
    for (int j = 0; j < 3; ++j){
      int wi = 2*wo - 1 + j; if ((unsigned)wi >= Ww) continue;
      for (int ci = 0; ci < 4; ++ci){
        float x = (ci < 3) ? img[(((size_t)b*3 + ci)*Hh + hi)*Ww + wi]
                           : sd[((size_t)b*Hh + hi)*Ww + wi];
        acc += x * w[((c*4 + ci)*3 + i)*3 + j];
      }
    }
  }
  float v = (acc + bc[c]) * (g[c]*rsqrtf(1.f+EPSF)) + bb[c];
  out[idx] = (_Float16)fmaxf(v, 0.f);
}

__global__ void k_small_conv(const _Float16* __restrict__ in, const float* __restrict__ w,
                             const float* __restrict__ bc, const float* __restrict__ g,
                             const float* __restrict__ bb, _Float16* __restrict__ out, int CC){
  int idx = blockIdx.x * blockDim.x + threadIdx.x;
  if (idx >= Bb * Hh2 * Ww2 * CC) return;
  int c = idx % CC; int q = idx / CC;
  int wo = q % Ww2; q /= Ww2;
  int ho = q % Hh2; int b = q / Hh2;
  float acc = 0.f;
  for (int i = 0; i < 3; ++i){
    int hi = ho - 1 + i; if ((unsigned)hi >= Hh2) continue;
    for (int j = 0; j < 3; ++j){
      int wi = wo - 1 + j; if ((unsigned)wi >= Ww2) continue;
      const _Float16* ip = in + ((size_t)(b*Hh2 + hi)*Ww2 + wi)*CC;
      const float* wp = w + (size_t)c*CC*9 + i*3 + j;
      for (int ci = 0; ci < CC; ++ci) acc += (float)ip[ci] * wp[ci*9];
    }
  }
  float v = (acc + bc[c]) * (g[c]*rsqrtf(1.f+EPSF)) + bb[c];
  out[idx] = (_Float16)fmaxf(v, 0.f);
}

__global__ void k_build_y(const _Float16* __restrict__ r2, const _Float16* __restrict__ st2,
                          _Float16* __restrict__ y){
  int idx = blockIdx.x*blockDim.x + threadIdx.x;
  if (idx >= Bb*Hh*Ww*CS) return;
  int c = idx & 63; int q = idx >> 6;
  int w = q % Ww; q /= Ww;
  int h = q % Hh; int b = q / Hh;
  int hs = h >> 1, wsx = w >> 1;
  _Float16 v = (_Float16)0.f;
  if (c < 32)      v = r2[((size_t)(b*Hh2 + hs)*Ww2 + wsx)*32 + c];
  else if (c < 48) v = st2[((size_t)(b*Hh2 + hs)*Ww2 + wsx)*16 + (c - 32)];
  y[idx] = v;
}

// ---------------------------------------------------------------- weight packing
// B-fragment per-lane order, chunk-major layout: wp[((ch*nct + ct)*32 + lane)*16 + e]
__global__ void k_cc_pack(const float* __restrict__ w, int cin, int cout,
                          int kch, int nct, _Float16* __restrict__ wp){
  int idx = blockIdx.x*blockDim.x + threadIdx.x;
  int total = nct*kch*32*16;
  if (idx >= total) return;
  int e = idx & 15; int q = idx >> 4;
  int lane = q & 31; q >>= 5;
  int ct = q % nct; int ch = q / nct;
  int k = ch*32 + ((e>>3)<<4) + ((lane>>4)<<3) + (e&7);
  int col = ct*16 + (lane & 15);
  float v = 0.f;
  if (k < cin && col < cout) v = w[(size_t)k*cout + col];
  wp[idx] = (_Float16)v;
}

__global__ void k_conv_pack(const float* __restrict__ w, const float* __restrict__ g,
                            int Cin, int OC, int kch, int nct, _Float16* __restrict__ wp){
  int idx = blockIdx.x*blockDim.x + threadIdx.x;
  int total = nct*kch*32*16;
  if (idx >= total) return;
  int e = idx & 15; int q = idx >> 4;
  int lane = q & 31; q >>= 5;
  int ct = q % nct; int ch = q / nct;
  int k = ch*32 + ((e>>3)<<4) + ((lane>>4)<<3) + (e&7);
  int col = ct*16 + (lane & 15);
  float v = 0.f;
  int ktot = 9*Cin;
  if (k < ktot && col < OC){
    int tap = k / Cin, c = k - tap*Cin;
    float s = g[col]*rsqrtf(1.f+EPSF);
    v = w[(((size_t)col*Cin + c)*3 + tap/3)*3 + (tap%3)] * s;
  }
  wp[idx] = (_Float16)v;
}

__global__ void k_fold_bias(const float* __restrict__ bc, const float* __restrict__ g,
                            const float* __restrict__ bb, int OC, float* __restrict__ out){
  int c = blockIdx.x*blockDim.x + threadIdx.x;
  if (c < OC) out[c] = bc[c]*(g[c]*rsqrtf(1.f+EPSF)) + bb[c];
}

// ---------------------------------------------------------------- 3D branch
__global__ void k_gather_fm(const _Float16* __restrict__ y, const int* __restrict__ pix,
                            _Float16* __restrict__ fm){
  int idx = blockIdx.x*blockDim.x + threadIdx.x;
  if (idx >= Bb*Nn*CS) return;
  int c = idx & 63; int q = idx >> 6;
  int n = q % Nn; int b = q / Nn;
  int p = pix[b*Nn + n];
  fm[idx] = y[((size_t)b*Hh*Ww + p)*CS + c];
}

// continuous conv: one wave per point n; tile rows = K neighbors.
// Rows 9..15 compute garbage from neighbor 0's (valid) row and are simply never
// read by the epilogue, so all loads are unconditional 16B vectors.
template<int C, int KCH, int NCT>
__global__ void k_cont_gemm_t(const _Float16* __restrict__ src, const float* __restrict__ coors,
                              const int* __restrict__ knn, const _Float16* __restrict__ wp,
                              const float* __restrict__ bias, _Float16* __restrict__ dst){
  __shared__ _Float16 ldsw[NCT*KCH*512];
  stage_weights(ldsw, wp, NCT*KCH*512);

  int wid = (blockIdx.x*blockDim.x + threadIdx.x) >> 5;    // grid sized exactly: Bb*Nn waves
  int lane = threadIdx.x & 31;
  int n = wid % Nn, b = wid / Nn;
  int m = lane & 15;
  int mm = (m < Kk) ? m : 0;                               // clamp: rows 9..15 read row of slot 0
  int khalf = (lane >> 4) << 3;
  int j = knn[((size_t)(b*Nn + n))*Kk + mm];
  const float* cn = coors + (size_t)(b*Nn + n)*3;
  const float* cj = coors + (size_t)(b*Nn + j)*3;
  float rx = cn[0]-cj[0], ry = cn[1]-cj[1], rz = cn[2]-cj[2];
  const _Float16* srow = src + ((size_t)b*Nn + j)*CS;

  v8f acc[NCT];
#pragma unroll
  for (int ct = 0; ct < NCT; ++ct)
#pragma unroll
    for (int r = 0; r < 8; ++r) acc[ct][r] = 0.f;

#pragma unroll
  for (int ch = 0; ch < KCH; ++ch){
    U16 au;
#pragma unroll
    for (int half = 0; half < 2; ++half){
      const int base = ch*32 + half*16;                    // compile-time; C % 16 == 0 =>
      if (base + 16 <= C){                                 // run either fully data or fully tail
        au.h[half] = *(const v8h*)(srow + base + khalf);   // aligned 16B vector load
      } else {
#pragma unroll
        for (int e = 0; e < 8; ++e){
          int k = base + khalf + e;
          float v = (k == C) ? rx : (k == C+1) ? ry : (k == C+2) ? rz : 0.f;
          au.e[half*8 + e] = (_Float16)v;
        }
      }
    }
#pragma unroll
    for (int ct = 0; ct < NCT; ++ct){
      v16h bm = *(const v16h*)(ldsw + ((size_t)(ch*NCT + ct)*32 + lane)*16);
      acc[ct] = __builtin_amdgcn_wmma_f32_16x16x32_f16(false, au.v, false, bm,
                                                       (short)0, acc[ct], false, false);
    }
  }
  int moff = (lane >> 4) << 3;
#pragma unroll
  for (int ct = 0; ct < NCT; ++ct){
    int col = ct*16 + (lane & 15);
    float bv = bias[col];
    float partial = 0.f;
#pragma unroll
    for (int r = 0; r < 8; ++r){
      if (r + moff < Kk) partial += fmaxf(acc[ct][r] + bv, 0.f);
    }
    partial += __shfl_xor(partial, 16, 32);   // fold M=8..15 half-lanes
    if (lane < 16) dst[((size_t)(b*Nn + n))*CS + col] = (_Float16)partial;
  }
}

__global__ void k_scatter(const _Float16* __restrict__ h2, const int* __restrict__ pix,
                          _Float16* __restrict__ t, int C){
  int idx = blockIdx.x*blockDim.x + threadIdx.x;
  if (idx >= Bb*Nn*CS) return;
  int c = idx & 63; if (c >= C) return;
  int q = idx >> 6; int n = q % Nn; int b = q / Nn;
  int p = pix[b*Nn + n];
  t[((size_t)b*Hh*Ww + p)*CS + c] = h2[idx];
}

// ---------------------------------------------------------------- 2D branch (depthwise)
__global__ void k_dw(const _Float16* __restrict__ in, int Hi, int Wi, int stride,
                     const float* __restrict__ w, const float* __restrict__ bc,
                     const float* __restrict__ g, const float* __restrict__ bb,
                     _Float16* __restrict__ out, int Ho, int Wo, int C){
  int idx = blockIdx.x*blockDim.x + threadIdx.x;
  if (idx >= Bb*Ho*Wo*CS) return;
  int c = idx & 63;
  if (c >= C){ out[idx] = (_Float16)0.f; return; }
  int q = idx >> 6; int wo = q % Wo; q /= Wo;
  int ho = q % Ho; int b = q / Ho;
  float acc = 0.f;
  for (int i = 0; i < 3; ++i){
    int hi = ho*stride - 1 + i; if ((unsigned)hi >= (unsigned)Hi) continue;
    for (int j = 0; j < 3; ++j){
      int wi = wo*stride - 1 + j; if ((unsigned)wi >= (unsigned)Wi) continue;
      acc += (float)in[((size_t)(b*Hi + hi)*Wi + wi)*CS + c] * w[c*9 + i*3 + j];
    }
  }
  float v = (acc + bc[c])*(g[c]*rsqrtf(1.f+EPSF)) + bb[c];
  out[idx] = (_Float16)fmaxf(v, 0.f);
}

__global__ void k_two_d_final(const _Float16* __restrict__ y, const _Float16* __restrict__ c2h,
                              const float* __restrict__ w3, const float* __restrict__ b3,
                              const float* __restrict__ g3, const float* __restrict__ bb3,
                              _Float16* __restrict__ t, int C){
  int idx = blockIdx.x*blockDim.x + threadIdx.x;
  if (idx >= Bb*Hh*Ww*CS) return;
  int c = idx & 63; if (c >= C) return;
  int q = idx >> 6; int w = q % Ww; q /= Ww;
  int h = q % Hh; int b = q / Hh;
  float acc = 0.f;
  for (int i = 0; i < 3; ++i){
    int hi = h - 1 + i; if ((unsigned)hi >= Hh) continue;
    for (int j = 0; j < 3; ++j){
      int wi = w - 1 + j; if ((unsigned)wi >= Ww) continue;
      acc += (float)y[((size_t)(b*Hh + hi)*Ww + wi)*CS + c] * w3[c*9 + i*3 + j];
    }
  }
  float c3 = fmaxf((acc + b3[c])*(g3[c]*rsqrtf(1.f+EPSF)) + bb3[c], 0.f);
  float c2v = (float)c2h[((size_t)(b*Hh2 + (h>>1))*Ww2 + (w>>1))*CS + c];
  t[idx] = (_Float16)((float)t[idx] + c2v + c3);
}

// ---------------------------------------------------------------- fusion conv (implicit GEMM, WMMA)
// One wave per 16-pixel row tile; computes all NCT 16-col output tiles (A built once per chunk).
// A runs are unconditional clamped 16B vector loads + register selects (no predicated loads).
template<int NCT>
__global__ void k_conv_gemm_t(const _Float16* __restrict__ tin, const _Float16* __restrict__ wp,
                              const float* __restrict__ bias, const _Float16* __restrict__ res,
                              _Float16* __restrict__ out, int Cin, int kch){
  __shared__ _Float16 ldsw[NCT*9*512];                     // max stage: NCT * (18/2) chunks
  int wid = (blockIdx.x*blockDim.x + threadIdx.x) >> 5;    // grid sized exactly: nrt waves
  int lane = threadIdx.x & 31;
  int row0 = wid << 4;                                     // 16 consecutive pixels, same row
  int b = row0 / (Hh*Ww); int hw = row0 % (Hh*Ww);
  int h = hw / Ww; int w0 = hw % Ww;
  int m = lane & 15;
  int khalf = (lane >> 4) << 3;
  int ktot = 9*Cin;
  __builtin_prefetch(tin + (size_t)row0*CS, 0, 1);

  v8f acc[NCT];
#pragma unroll
  for (int ct = 0; ct < NCT; ++ct)
#pragma unroll
    for (int r = 0; r < 8; ++r) acc[ct][r] = 0.f;

  int stage = kch >> 1;
  for (int s = 0; s < 2; ++s){
    stage_weights(ldsw, wp + (size_t)s*stage*NCT*512, stage*NCT*512);
    for (int cl = 0; cl < stage; ++cl){
      int ch = s*stage + cl;
      U16 au;
#pragma unroll
      for (int half = 0; half < 2; ++half){
        int k0 = ch*32 + half*16 + khalf;        // run of 8 stays inside one tap (Cin % 8 == 0)
        int tap = k0 / Cin; int c0 = k0 - tap*Cin;
        int hh  = h + tap/3 - 1;
        int wwp = w0 + m + (tap%3) - 1;
        bool ok = (k0 < ktot) && ((unsigned)hh < Hh) && ((unsigned)wwp < Ww);
        int hc = hh < 0 ? 0 : (hh >= Hh ? Hh-1 : hh);          // clamped, always-valid address
        int wc = wwp < 0 ? 0 : (wwp >= Ww ? Ww-1 : wwp);
        v8h lo = *(const v8h*)(tin + ((size_t)(b*Hh + hc)*Ww + wc)*CS + c0);
#pragma unroll
        for (int e = 0; e < 8; ++e)
          au.e[half*8 + e] = ok ? lo[e] : (_Float16)0.f;        // v_cndmask, no exec dance
      }
#pragma unroll
      for (int ct = 0; ct < NCT; ++ct){
        v16h bm = *(const v16h*)(ldsw + ((size_t)(cl*NCT + ct)*32 + lane)*16);
        acc[ct] = __builtin_amdgcn_wmma_f32_16x16x32_f16(false, au.v, false, bm,
                                                         (short)0, acc[ct], false, false);
      }
    }
  }
  int moff = (lane >> 4) << 3;
#pragma unroll
  for (int ct = 0; ct < NCT; ++ct){
    int col = ct*16 + (lane & 15);
    float bv = bias[col];
#pragma unroll
    for (int r = 0; r < 8; ++r){
      int row = row0 + r + moff;
      float v = fmaxf(acc[ct][r] + bv, 0.f);
      size_t o = (size_t)row*CS + col;
      if (res) v += (float)res[o];
      out[o] = (_Float16)v;
    }
  }
}

// ---------------------------------------------------------------- head final 32->1 conv + relu
__global__ void k_head2(const _Float16* __restrict__ hb, const float* __restrict__ w,
                        const float* __restrict__ bc, float* __restrict__ out){
  int idx = blockIdx.x*blockDim.x + threadIdx.x;
  if (idx >= Bb*Hh*Ww) return;
  int q = idx; int x = q % Ww; q /= Ww;
  int h = q % Hh; int b = q / Hh;
  float acc = 0.f;
  for (int i = 0; i < 3; ++i){
    int hi = h - 1 + i; if ((unsigned)hi >= Hh) continue;
    for (int j = 0; j < 3; ++j){
      int wi = x - 1 + j; if ((unsigned)wi >= Ww) continue;
      const _Float16* p = hb + ((size_t)(b*Hh + hi)*Ww + wi)*CS;
      for (int c = 0; c < 32; ++c) acc += (float)p[c] * w[(c*3 + i)*3 + j];
    }
  }
  out[idx] = fmaxf(acc + bc[0], 0.f);
}

// ================================================================ host
extern "C" void kernel_launch(void* const* d_in, const int* in_sizes, int n_in,
                              void* d_out, int out_size, void* d_ws, size_t ws_size,
                              hipStream_t stream){
  (void)in_sizes; (void)n_in; (void)out_size; (void)ws_size;
  const float* img = (const float*)d_in[0];
  const float* sd  = (const float*)d_in[1];
  const unsigned char* mask = (const unsigned char*)d_in[2];
  const float* coors = (const float*)d_in[3];
  const int*   knn   = (const int*)d_in[4];
  void* const* P = d_in + 5;
  // params flattened in jax pytree order (dict keys sorted alphabetically):
  // blocks[12] x {bn1.b,bn1.g,bn2.b,bn2.g,bn3.b,bn3.g,cc1.b,cc1.w,cc2.b,cc2.w,
  //               dw1.b,dw1.w,dw2.b,dw2.w,dw3.b,dw3.w,out_bn.b,out_bn.g,out_conv.b,out_conv.w}
  // head: [240..245] bn.b,bn.g,conv0.b,conv0.w,conv1.b,conv1.w
  // rgbd: [246..253] bn0.b,bn0.g,conv0.b,conv0.w,bn1.b,bn1.g,conv1.b,conv1.w
  // sparse: [254..261] same layout as rgbd
#define F(i) ((const float*)P[(i)])

  size_t off = 0;
  auto alloc = [&](size_t bytes)->char*{
    char* p = (char*)d_ws + off;
    off += (bytes + 255) & ~(size_t)255;
    return p;
  };
  _Float16* y   = (_Float16*)alloc((size_t)Bb*Hh*Ww*CS*sizeof(_Float16));
  _Float16* y2  = (_Float16*)alloc((size_t)Bb*Hh*Ww*CS*sizeof(_Float16));
  _Float16* t   = (_Float16*)alloc((size_t)Bb*Hh*Ww*CS*sizeof(_Float16));
  _Float16* c1h = (_Float16*)alloc((size_t)Bb*Hh2*Ww2*CS*sizeof(_Float16));
  _Float16* c2h = (_Float16*)alloc((size_t)Bb*Hh2*Ww2*CS*sizeof(_Float16));
  _Float16* fm  = (_Float16*)alloc((size_t)Bb*Nn*CS*sizeof(_Float16));
  _Float16* h1  = (_Float16*)alloc((size_t)Bb*Nn*CS*sizeof(_Float16));
  _Float16* h2  = (_Float16*)alloc((size_t)Bb*Nn*CS*sizeof(_Float16));
  int*      pix = (int*)alloc((size_t)Bb*Nn*sizeof(int));
  _Float16* wpkA  = (_Float16*)alloc((size_t)4*18*32*16*sizeof(_Float16));
  _Float16* wpkC1 = (_Float16*)alloc((size_t)4*3*32*16*sizeof(_Float16));
  _Float16* wpkC2 = (_Float16*)alloc((size_t)4*3*32*16*sizeof(_Float16));
  _Float16* wpkH  = (_Float16*)alloc((size_t)2*18*32*16*sizeof(_Float16));
  float* biasA = (float*)alloc(64*sizeof(float));
  float* biasH = (float*)alloc(32*sizeof(float));

  const int TB = 256;
  auto nb = [](long long n)->unsigned{ return (unsigned)((n + 255) / 256); };

  // --- pix_idx (order-preserving nonzero, size N, fill 0)
  k_pix_idx<<<Bb, 1, 0, stream>>>(mask, pix);

  // --- stems (reuse big scratch buffers before block loop)
  _Float16* st1 = c1h; _Float16* st2 = c2h;
  _Float16* r1  = t;   _Float16* r2  = y2;
  k_sparse1<<<nb((long long)Bb*Hh2*Ww2*16), TB, 0, stream>>>(sd, F(257), F(256), F(255), F(254), st1);
  k_small_conv<<<nb((long long)Bb*Hh2*Ww2*16), TB, 0, stream>>>(st1, F(261), F(260), F(259), F(258), st2, 16);
  k_rgbd1<<<nb((long long)Bb*Hh2*Ww2*32), TB, 0, stream>>>(img, sd, F(249), F(248), F(247), F(246), r1);
  k_small_conv<<<nb((long long)Bb*Hh2*Ww2*32), TB, 0, stream>>>(r1, F(253), F(252), F(251), F(250), r2, 32);
  k_build_y<<<nb((long long)Bb*Hh*Ww*CS), TB, 0, stream>>>(r2, st2, y);

  const unsigned contBlocks = (Bb*Nn) / 8;            // 2048: exact, no early-exit waves
  const unsigned convBlocks = (Bb*Hh*Ww/16) / 8;      // 2048: exact, no early-exit waves

  // --- 12 fusion blocks
  for (int i = 0; i < 12; ++i){
    int base = i*20;
    const float *bn1b=F(base+0), *bn1g=F(base+1), *bn2b=F(base+2), *bn2g=F(base+3),
                *bn3b=F(base+4), *bn3g=F(base+5), *cc1b=F(base+6), *cc1w=F(base+7),
                *cc2b=F(base+8), *cc2w=F(base+9), *dw1b=F(base+10), *dw1w=F(base+11),
                *dw2b=F(base+12), *dw2w=F(base+13), *dw3b=F(base+14), *dw3w=F(base+15),
                *obnb=F(base+16), *obng=F(base+17), *ocb=F(base+18), *ocw=F(base+19);
    int C = (i == 0) ? 48 : 64;
    int kcc = (C + 3 + 31) / 32;     // 2 (C=48) or 3 (C=64)
    int nctc = C / 16;               // 3 or 4
    int kco = (9*C + 31) / 32;       // 14 or 18 (even -> two LDS stages)

    // weight packing (BN folded for out_conv)
    k_cc_pack<<<nb((long long)nctc*kcc*512), TB, 0, stream>>>(cc1w, C+3, C, kcc, nctc, wpkC1);
    k_cc_pack<<<nb((long long)nctc*kcc*512), TB, 0, stream>>>(cc2w, C+3, C, kcc, nctc, wpkC2);
    k_conv_pack<<<nb((long long)4*kco*512), TB, 0, stream>>>(ocw, obng, C, 64, kco, 4, wpkA);
    k_fold_bias<<<1, 64, 0, stream>>>(ocb, obng, obnb, 64, biasA);

    // 3D branch: gather -> contconv -> contconv -> scatter into zeroed t
    k_gather_fm<<<nb((long long)Bb*Nn*CS), TB, 0, stream>>>(y, pix, fm);
    if (i == 0){
      k_cont_gemm_t<48,2,3><<<contBlocks, TB, 0, stream>>>(fm, coors, knn, wpkC1, cc1b, h1);
      k_cont_gemm_t<48,2,3><<<contBlocks, TB, 0, stream>>>(h1, coors, knn, wpkC2, cc2b, h2);
    } else {
      k_cont_gemm_t<64,3,4><<<contBlocks, TB, 0, stream>>>(fm, coors, knn, wpkC1, cc1b, h1);
      k_cont_gemm_t<64,3,4><<<contBlocks, TB, 0, stream>>>(h1, coors, knn, wpkC2, cc2b, h2);
    }
    k_zero_h<<<4096, TB, 0, stream>>>(t, (size_t)Bb*Hh*Ww*CS);
    k_scatter<<<nb((long long)Bb*Nn*CS), TB, 0, stream>>>(h2, pix, t, C);

    // 2D branch fused into t
    k_dw<<<nb((long long)Bb*Hh2*Ww2*CS), TB, 0, stream>>>(y, Hh, Ww, 2, dw1w, dw1b, bn1g, bn1b, c1h, Hh2, Ww2, C);
    k_dw<<<nb((long long)Bb*Hh2*Ww2*CS), TB, 0, stream>>>(c1h, Hh2, Ww2, 1, dw2w, dw2b, bn2g, bn2b, c2h, Hh2, Ww2, C);
    k_two_d_final<<<nb((long long)Bb*Hh*Ww*CS), TB, 0, stream>>>(y, c2h, dw3w, dw3b, bn3g, bn3b, t, C);

    // fusion conv (WMMA implicit GEMM) + BN + ReLU + residual (blocks 1..11)
    const _Float16* resp = (i > 0) ? (const _Float16*)y : (const _Float16*)nullptr;
    k_conv_gemm_t<4><<<convBlocks, TB, 0, stream>>>(t, wpkA, biasA, resp, y2, C, kco);
    _Float16* tmp = y; y = y2; y2 = tmp;
  }

  // --- head: 64->32 WMMA conv (+BN+ReLU), then 32->1 VALU conv (+ReLU) to f32 output
  k_conv_pack<<<nb((long long)2*18*512), TB, 0, stream>>>(F(243), F(241), 64, 32, 18, 2, wpkH);
  k_fold_bias<<<1, 32, 0, stream>>>(F(242), F(241), F(240), 32, biasH);
  k_conv_gemm_t<2><<<convBlocks, TB, 0, stream>>>(y, wpkH, biasH, nullptr, t, 64, 18);
  k_head2<<<nb((long long)Bb*Hh*Ww), TB, 0, stream>>>(t, F(245), F(244), (float*)d_out);
#undef F
}
